// SimpleRNN_34883724378310
// MI455X (gfx1250) — matrix operand order repaired and checked by reading, not compiled, not run on hardware
//
#include <hip/hip_runtime.h>
#include <hip/hip_bf16.h>
#include <stdint.h>

typedef __bf16 bf16;
typedef __attribute__((ext_vector_type(16))) __bf16 v16bf;
typedef __attribute__((ext_vector_type(8)))  __bf16 v8bf;
typedef __attribute__((ext_vector_type(8)))  float  v8f;
typedef __attribute__((ext_vector_type(4)))  unsigned int u32x4;
typedef __attribute__((ext_vector_type(8)))  int i32x8;
typedef __attribute__((ext_vector_type(4)))  int i32x4;

union AFrag { v16bf v; v8bf h[2]; };

#define BB 64
#define SS 512
#define DD 512
#define HH 512
#define VV 32000
#define CC 1000
#define FFN 1024

#if defined(__has_builtin)
#if __has_builtin(__builtin_amdgcn_tensor_load_to_lds) && __has_builtin(__builtin_amdgcn_s_wait_tensorcnt)
#define HAVE_TDM 1
#endif
#if __has_builtin(__builtin_amdgcn_sched_barrier)
#define SCHED_FENCE() __builtin_amdgcn_sched_barrier(0)
#endif
#endif
#ifndef HAVE_TDM
#define HAVE_TDM 0
#endif
#ifndef SCHED_FENCE
#define SCHED_FENCE()
#endif

// ---------------------------------------------------------------------------
// Prep: f32 -> bf16 elementwise (embedding table)
// ---------------------------------------------------------------------------
__global__ void f32_to_bf16_kernel(const float* __restrict__ src,
                                   bf16* __restrict__ dst, long n) {
  long i = (long)blockIdx.x * blockDim.x + threadIdx.x;
  if (i < n) dst[i] = (bf16)src[i];
}

// ---------------------------------------------------------------------------
// Prep: swizzle a row-major f32 weight [Ksrc,Nsrc] into WMMA B-fragment order
// (bf16, padded to Kp x Np).  For tile (kt,nt), lane L holds N = nt*16+(L&15),
// K = kt*32 + 16*(L>>4) + i, i = 0..15, stored contiguous (32B per lane).
// ---------------------------------------------------------------------------
__global__ void swizzle_b_kernel(const float* __restrict__ src,
                                 bf16* __restrict__ dst,
                                 int Kp, int Np, int Ksrc, int Nsrc) {
  int NT = Np >> 4;
  int total = (Kp >> 5) * NT * 32;
  int id = blockIdx.x * blockDim.x + threadIdx.x;
  if (id >= total) return;
  int lane = id & 31;
  int frag = id >> 5;
  int nt = frag % NT;
  int kt = frag / NT;
  int n = nt * 16 + (lane & 15);
  int kbase = kt * 32 + ((lane >> 4) << 4);
  bf16* out = dst + (size_t)id * 16;
#pragma unroll
  for (int i = 0; i < 16; i++) {
    int k = kbase + i;
    float v = (k < Ksrc && n < Nsrc) ? src[(size_t)k * Nsrc + n] : 0.f;
    out[i] = (bf16)v;
  }
}

// ---------------------------------------------------------------------------
// xproj[m, :] = emb[tokens[m], :] @ Wx + b_rnn     (m = b*S + s, M = 32768)
// Software-pipelined k-loop: stage kt+1 fragments load while stage kt WMMAs
// execute (separate registers + sched fence -> nonzero loadcnt waits).
// ---------------------------------------------------------------------------
__global__ void xproj_kernel(const int* __restrict__ tokens,
                             const bf16* __restrict__ embb,
                             const bf16* __restrict__ wxswz,
                             const float* __restrict__ b_rnn,
                             float* __restrict__ xproj) {
  const int NT = HH >> 4;   // 32
  const int KT = DD >> 5;   // 16
  int lane = threadIdx.x & 31;
  int wave = threadIdx.x >> 5;
  int half = lane >> 4;
  int mtile = blockIdx.x;
  int mrow = mtile * 16 + (lane & 15);
  int tok = tokens[mrow];
  const bf16* arow = embb + (size_t)tok * DD;
  int nt0 = wave * 4;

  v8f acc[4];
#pragma unroll
  for (int j = 0; j < 4; j++)
#pragma unroll
    for (int e = 0; e < 8; e++) acc[j][e] = 0.f;

  AFrag a_cur, a_nxt;
  v16bf b_cur[4], b_nxt[4];
  a_cur.h[0] = *(const v8bf*)(arow + 16 * half);
  a_cur.h[1] = *(const v8bf*)(arow + 16 * half + 8);
#pragma unroll
  for (int j = 0; j < 4; j++)
    b_cur[j] = *(const v16bf*)(wxswz + ((size_t)(nt0 + j) * 32 + lane) * 16);

#pragma unroll
  for (int kt = 0; kt < KT; kt++) {
    if (kt + 1 < KT) {
      int k1 = (kt + 1) * 32;
      a_nxt.h[0] = *(const v8bf*)(arow + k1 + 16 * half);
      a_nxt.h[1] = *(const v8bf*)(arow + k1 + 16 * half + 8);
#pragma unroll
      for (int j = 0; j < 4; j++)
        b_nxt[j] = *(const v16bf*)(wxswz + ((size_t)((kt + 1) * NT + nt0 + j) * 32 + lane) * 16);
    }
    SCHED_FENCE();
#pragma unroll
    for (int j = 0; j < 4; j++)
      acc[j] = __builtin_amdgcn_wmma_f32_16x16x32_bf16(
          false, a_cur.v, false, b_cur[j], (short)0, acc[j], false, false);
    SCHED_FENCE();
    a_cur = a_nxt;
#pragma unroll
    for (int j = 0; j < 4; j++) b_cur[j] = b_nxt[j];
  }

#pragma unroll
  for (int j = 0; j < 4; j++) {
    int n = (nt0 + j) * 16 + (lane & 15);
    float bias = b_rnn[n];
#pragma unroll
    for (int v = 0; v < 8; v++) {
      int r = mtile * 16 + v + 8 * half;
      xproj[(size_t)r * HH + n] = acc[j][v] + bias;
    }
  }
}

// ---------------------------------------------------------------------------
// TDM issue: 2-D tile load, 16 rows x 512 f32, row stride S*H floats.
// D# per CDNA5 ISA (group0 128b, group1 256b).
// ---------------------------------------------------------------------------
#if HAVE_TDM
__device__ __forceinline__ void tdm_load_xproj_tile(const float* gsrc,
                                                    unsigned lds_off) {
  unsigned long long ga = (unsigned long long)(uintptr_t)gsrc;
  u32x4 g0;
  g0[0] = 1u;                                            // count=1, user D#
  g0[1] = lds_off;                                       // lds_addr (bytes)
  g0[2] = (unsigned)ga;                                  // global_addr[31:0]
  g0[3] = (unsigned)((ga >> 32) & 0x01FFFFFFu) | (2u << 30); // [56:32], type=2
  i32x8 g1;
  g1[0] = (int)(2u << 16);        // workgroup_mask=0, data_size=2 (4 bytes)
  g1[1] = (int)(512u << 16);      // tensor_dim0 = 512 (bits 79:48)
  g1[2] = (int)(16u << 16);       // tensor_dim1 = 16  (bits 111:80)
  g1[3] = (int)(512u << 16);      // tile_dim0 = 512   (bits 127:112)
  g1[4] = 16;                     // tile_dim1 = 16, tile_dim2 = 0
  g1[5] = (int)(SS * HH);         // tensor_dim0_stride = 262144 elements
  g1[6] = 0;                      // tensor_dim1_stride = 0 (2-D)
  g1[7] = 0;
  i32x4 gz = {0, 0, 0, 0};
#if __clang_major__ >= 23
  i32x8 gz8 = {0, 0, 0, 0, 0, 0, 0, 0};
  __builtin_amdgcn_tensor_load_to_lds(g0, g1, gz, gz, gz8, 0);
#else
  __builtin_amdgcn_tensor_load_to_lds(g0, g1, gz, gz, 0);
#endif
}
#endif

// ---------------------------------------------------------------------------
// RNN scan: 4 blocks, each owns 16 batch rows for all 512 steps.
// h (bf16) in LDS; f32 carry in registers; xproj slice for the next step
// DMA'd into LDS by the TDM under that step's 64 WMMAs; token masks
// packed into an LDS bitset once.  k-loop software-pipelined as in xproj.
// ---------------------------------------------------------------------------
#define LDSTR 528   // 512 + 16 bf16 pad: 1056B row stride breaks bank aliasing

__global__ void rnn_scan_kernel(const int* __restrict__ tokens,
                                const float* __restrict__ xproj,
                                const bf16* __restrict__ whswz,
                                bf16* __restrict__ hout) {
  const int NT = HH >> 4;   // 32
  const int KT = HH >> 5;   // 16
  __shared__ bf16 hb[16 * LDSTR];          // 16.5 KB
  __shared__ unsigned mbits[16 * 16];      // 1 KB: 16 rows x 512 mask bits
#if HAVE_TDM
  __shared__ float xp[16 * HH];            // 32 KB TDM landing buffer
#endif
  int lane = threadIdx.x & 31;
  int wave = threadIdx.x >> 5;
  int half = lane >> 4;
  int b0 = blockIdx.x * 16;
  int nt0 = wave * 4;

  for (int i = threadIdx.x; i < 16 * LDSTR; i += blockDim.x) hb[i] = (bf16)0.f;
  {
    int row = threadIdx.x >> 4;            // 0..15
    int w   = threadIdx.x & 15;            // 0..15 words of 32 bits
    unsigned bits = 0;
#pragma unroll
    for (int i = 0; i < 32; i++)
      bits |= (tokens[(b0 + row) * SS + w * 32 + i] != 0 ? 1u : 0u) << i;
    mbits[row * 16 + w] = bits;
  }

#if HAVE_TDM
  unsigned xp_off = (unsigned)(uintptr_t)(&xp[0]);   // flat->LDS low 32 bits
  if (wave == 0)
    tdm_load_xproj_tile(xproj + (size_t)(b0 * SS + 0) * HH, xp_off);
#endif
  __syncthreads();

  float hprev[4][8];
#pragma unroll
  for (int j = 0; j < 4; j++)
#pragma unroll
    for (int v = 0; v < 8; v++) hprev[j][v] = 0.f;

  const bf16* arow = hb + (size_t)(lane & 15) * LDSTR;

  for (int s = 0; s < SS; s++) {
    v8f acc[4];
#pragma unroll
    for (int j = 0; j < 4; j++)
#pragma unroll
      for (int e = 0; e < 8; e++) acc[j][e] = 0.f;

    AFrag a_cur, a_nxt;
    v16bf b_cur[4], b_nxt[4];
    a_cur.h[0] = *(const v8bf*)(arow + 16 * half);
    a_cur.h[1] = *(const v8bf*)(arow + 16 * half + 8);
#pragma unroll
    for (int j = 0; j < 4; j++)
      b_cur[j] = *(const v16bf*)(whswz + ((size_t)(nt0 + j) * 32 + lane) * 16);

#pragma unroll
    for (int kt = 0; kt < KT; kt++) {
      if (kt + 1 < KT) {
        int k1 = (kt + 1) * 32;
        a_nxt.h[0] = *(const v8bf*)(arow + k1 + 16 * half);
        a_nxt.h[1] = *(const v8bf*)(arow + k1 + 16 * half + 8);
#pragma unroll
        for (int j = 0; j < 4; j++)
          b_nxt[j] = *(const v16bf*)(whswz + ((size_t)((kt + 1) * NT + nt0 + j) * 32 + lane) * 16);
      }
      SCHED_FENCE();
#pragma unroll
      for (int j = 0; j < 4; j++)
        acc[j] = __builtin_amdgcn_wmma_f32_16x16x32_bf16(
            false, a_cur.v, false, b_cur[j], (short)0, acc[j], false, false);
      SCHED_FENCE();
      a_cur = a_nxt;
#pragma unroll
      for (int j = 0; j < 4; j++) b_cur[j] = b_nxt[j];
    }
    __syncthreads();   // all waves done reading hb for step s

#if HAVE_TDM
    if (wave == 0) __builtin_amdgcn_s_wait_tensorcnt(0);  // xp[s] landed
    __syncthreads();   // xp visible to all waves
#endif

    unsigned mword = (unsigned)(s >> 5);
    unsigned mbit  = (unsigned)(s & 31);
#pragma unroll
    for (int j = 0; j < 4; j++) {
      int n = (nt0 + j) * 16 + (lane & 15);
#pragma unroll
      for (int v = 0; v < 8; v++) {
        int row = v + 8 * half;
#if HAVE_TDM
        float xv = xp[row * HH + n];
#else
        float xv = xproj[(size_t)((b0 + row) * SS + s) * HH + n];
#endif
        float nv = fmaxf(acc[j][v] + xv, 0.f);
        bool live = (mbits[row * 16 + mword] >> mbit) & 1u;
        float keep = live ? nv : hprev[j][v];
        hprev[j][v] = keep;
        hb[row * LDSTR + n] = (bf16)keep;
      }
    }
    __syncthreads();   // hb writes + xp reads complete

#if HAVE_TDM
    if (wave == 0 && s + 1 < SS)   // DMA next step's slice under this step
      tdm_load_xproj_tile(xproj + (size_t)(b0 * SS + (s + 1)) * HH, xp_off);
#endif
  }

#pragma unroll
  for (int j = 0; j < 4; j++) {
    int n = (nt0 + j) * 16 + (lane & 15);
#pragma unroll
    for (int v = 0; v < 8; v++) {
      int row = v + 8 * half;
      hout[(size_t)(b0 + row) * HH + n] = (bf16)hprev[j][v];
    }
  }
}

// ---------------------------------------------------------------------------
// MLP head GEMM: out[M=64, N] = act(A[64,K] @ Bswz + bias), pipelined k-loop.
// ---------------------------------------------------------------------------
__global__ void gemm_head_kernel(const bf16* __restrict__ A,
                                 const bf16* __restrict__ Bswz,
                                 const float* __restrict__ bias, int nbias,
                                 bf16* __restrict__ outb,
                                 float* __restrict__ outf,
                                 int K, int N, int relu) {
  int NT = N >> 4;
  int KT = K >> 5;
  int lane = threadIdx.x & 31;
  int wave = threadIdx.x >> 5;
  int half = lane >> 4;
  int mtile = blockIdx.x;
  int nt0 = blockIdx.y * 32 + wave * 4;
  const bf16* arow = A + (size_t)(mtile * 16 + (lane & 15)) * K;

  v8f acc[4];
#pragma unroll
  for (int j = 0; j < 4; j++)
#pragma unroll
    for (int e = 0; e < 8; e++) acc[j][e] = 0.f;

  AFrag a_cur, a_nxt;
  v16bf b_cur[4], b_nxt[4];
  a_cur.h[0] = *(const v8bf*)(arow + 16 * half);
  a_cur.h[1] = *(const v8bf*)(arow + 16 * half + 8);
#pragma unroll
  for (int j = 0; j < 4; j++)
    b_cur[j] = *(const v16bf*)(Bswz + ((size_t)(nt0 + j) * 32 + lane) * 16);

  for (int kt = 0; kt < KT; kt++) {
    if (kt + 1 < KT) {
      int k1 = (kt + 1) * 32;
      a_nxt.h[0] = *(const v8bf*)(arow + k1 + 16 * half);
      a_nxt.h[1] = *(const v8bf*)(arow + k1 + 16 * half + 8);
#pragma unroll
      for (int j = 0; j < 4; j++)
        b_nxt[j] = *(const v16bf*)(Bswz + ((size_t)((kt + 1) * NT + nt0 + j) * 32 + lane) * 16);
    }
    SCHED_FENCE();
#pragma unroll
    for (int j = 0; j < 4; j++)
      acc[j] = __builtin_amdgcn_wmma_f32_16x16x32_bf16(
          false, a_cur.v, false, b_cur[j], (short)0, acc[j], false, false);
    SCHED_FENCE();
    a_cur = a_nxt;
#pragma unroll
    for (int j = 0; j < 4; j++) b_cur[j] = b_nxt[j];
  }

#pragma unroll
  for (int j = 0; j < 4; j++) {
    int n = (nt0 + j) * 16 + (lane & 15);
    float bv = (n < nbias) ? bias[n] : 0.f;
#pragma unroll
    for (int v = 0; v < 8; v++) {
      int row = mtile * 16 + v + 8 * half;
      float val = acc[j][v] + bv;
      if (relu) val = fmaxf(val, 0.f);
      if (outb) outb[(size_t)row * N + n] = (bf16)val;
      else      outf[(size_t)row * N + n] = val;
    }
  }
}

// ---------------------------------------------------------------------------
// Row softmax over first 1000 of 1024 padded logits. One block per row.
// ---------------------------------------------------------------------------
__global__ void softmax_kernel(const float* __restrict__ logits,
                               float* __restrict__ out) {
  const int LP = 1024;
  __shared__ float red[256];
  int row = blockIdx.x;
  int tid = threadIdx.x;

  float mx = -3.402823466e38f;
  for (int n = tid; n < CC; n += 256) mx = fmaxf(mx, logits[(size_t)row * LP + n]);
  red[tid] = mx;
  __syncthreads();
  for (int s = 128; s > 0; s >>= 1) {
    if (tid < s) red[tid] = fmaxf(red[tid], red[tid + s]);
    __syncthreads();
  }
  mx = red[0];
  __syncthreads();

  float se = 0.f;
  for (int n = tid; n < CC; n += 256) se += expf(logits[(size_t)row * LP + n] - mx);
  red[tid] = se;
  __syncthreads();
  for (int s = 128; s > 0; s >>= 1) {
    if (tid < s) red[tid] += red[tid + s];
    __syncthreads();
  }
  se = red[0];

  float inv = 1.f / se;
  for (int n = tid; n < CC; n += 256)
    out[(size_t)row * CC + n] = expf(logits[(size_t)row * LP + n] - mx) * inv;
}

// ---------------------------------------------------------------------------
extern "C" void kernel_launch(void* const* d_in, const int* in_sizes, int n_in,
                              void* d_out, int out_size, void* d_ws, size_t ws_size,
                              hipStream_t stream) {
  (void)in_sizes; (void)n_in; (void)out_size; (void)ws_size;

  const int*   tokens = (const int*)  d_in[0];
  const float* emb    = (const float*)d_in[1];
  const float* Wx     = (const float*)d_in[2];
  const float* Wh     = (const float*)d_in[3];
  const float* b_rnn  = (const float*)d_in[4];
  const float* W1     = (const float*)d_in[5];
  const float* b1     = (const float*)d_in[6];
  const float* W2     = (const float*)d_in[7];
  const float* b2     = (const float*)d_in[8];
  const float* Wo     = (const float*)d_in[9];
  const float* bo     = (const float*)d_in[10];
  float* out = (float*)d_out;

  char* base = (char*)d_ws;
  size_t off = 0;
  auto alloc = [&](size_t bytes) -> char* {
    char* p = base + off;
    off = (off + bytes + 255) & ~(size_t)255;
    return p;
  };
  bf16*  embb   = (bf16*) alloc((size_t)VV * DD * 2);       // 32.8 MB
  bf16*  wxswz  = (bf16*) alloc((size_t)DD * HH * 2);
  bf16*  whswz  = (bf16*) alloc((size_t)HH * HH * 2);
  bf16*  w1swz  = (bf16*) alloc((size_t)HH * FFN * 2);
  bf16*  w2swz  = (bf16*) alloc((size_t)FFN * FFN * 2);
  bf16*  woswz  = (bf16*) alloc((size_t)FFN * 1024 * 2);    // N padded 1000->1024
  float* xproj  = (float*)alloc((size_t)BB * SS * HH * 4);  // 67 MB
  bf16*  hout   = (bf16*) alloc((size_t)BB * HH * 2);
  bf16*  y1     = (bf16*) alloc((size_t)BB * FFN * 2);
  bf16*  y2     = (bf16*) alloc((size_t)BB * FFN * 2);
  float* logits = (float*)alloc((size_t)BB * 1024 * 4);

  // --- prep: bf16 conversion + B-fragment swizzles ---
  {
    long n = (long)VV * DD;
    f32_to_bf16_kernel<<<dim3((unsigned)((n + 255) / 256)), dim3(256), 0, stream>>>(emb, embb, n);
  }
  auto swz = [&](const float* src, bf16* dst, int Kp, int Np, int Ksrc, int Nsrc) {
    int total = (Kp >> 5) * (Np >> 4) * 32;
    swizzle_b_kernel<<<dim3((total + 255) / 256), dim3(256), 0, stream>>>(src, dst, Kp, Np, Ksrc, Nsrc);
  };
  swz(Wx, wxswz, DD,  HH,   DD,  HH);
  swz(Wh, whswz, HH,  HH,   HH,  HH);
  swz(W1, w1swz, HH,  FFN,  HH,  FFN);
  swz(W2, w2swz, FFN, FFN,  FFN, FFN);
  swz(Wo, woswz, FFN, 1024, FFN, CC);

  // --- xproj GEMM: 2048 M-tiles ---
  xproj_kernel<<<dim3((BB * SS) / 16), dim3(256), 0, stream>>>(tokens, embb, wxswz, b_rnn, xproj);

  // --- serial scan: 4 blocks, each owns 16 batch rows ---
  rnn_scan_kernel<<<dim3(BB / 16), dim3(256), 0, stream>>>(tokens, xproj, whswz, hout);

  // --- MLP head ---
  gemm_head_kernel<<<dim3(BB / 16, FFN / 512), dim3(256), 0, stream>>>(
      hout, w1swz, b1, FFN, y1, nullptr, HH, FFN, 1);
  gemm_head_kernel<<<dim3(BB / 16, FFN / 512), dim3(256), 0, stream>>>(
      y1, w2swz, b2, FFN, y2, nullptr, FFN, FFN, 1);
  gemm_head_kernel<<<dim3(BB / 16, 1024 / 512), dim3(256), 0, stream>>>(
      y2, woswz, bo, CC, nullptr, logits, FFN, 1024, 0);

  // --- softmax ---
  softmax_kernel<<<dim3(BB), dim3(256), 0, stream>>>(logits, out);
}
